// TwoRegionRNNBase_48455821033983
// MI455X (gfx1250) — matrix operand order repaired and checked by reading, not compiled, not run on hardware
//
#include <hip/hip_runtime.h>

typedef __attribute__((ext_vector_type(16))) _Float16 v16h;
typedef __attribute__((ext_vector_type(8)))  float    v8f;
typedef __attribute__((ext_vector_type(4)))  float    v4f;

#define T_STEPS 256
#define BATCH   256
#define DIN     32
#define HID     256
#define OUT_O   8
#define ALPHA_A 0.1f
#define ALPHA_B 0.06666666666666667f

// half-element offsets of each staged section inside d_ws
#define OFF_AA   0
#define OFF_AB   65536
#define OFF_BB   131072
#define OFF_BA   196608
#define OFF_INA  262144
#define OFF_INB  270336
#define OFF_OUTW 278528

__device__ __forceinline__ v8f wmma16(v16h a, v16h b, v8f c) {
  // D = A(16x32 f16) * B(32x16 f16) + C(16x16 f32)
  return __builtin_amdgcn_wmma_f32_16x16x32_f16(false, a, false, b, (short)0, c,
                                                false, false);
}

// Stage a weight matrix W[N,K] (row-major f32) as W^T in WMMA B-operand
// fragment order, f16.  Fragment (kc,nt) holds B[k= kc*32+kl][n= nt*16+nl]
// with lane = ((kl>>3)&1)*16 + nl and per-lane halves h: kl = (h>>3)*16 + (lane>>4)*8 + (h&7).
// dst index = ((kc*NT + nt)*32 + lane)*16 + h.  Optional per-k mask folded in.
__global__ void stage_weights_kernel(const float* __restrict__ src,
                                     _Float16* __restrict__ dst,
                                     int Nreal, int Kreal, int NT, int KC,
                                     int Kstride, const float* __restrict__ mask) {
  int idx = blockIdx.x * blockDim.x + threadIdx.x;
  int total = NT * KC * 512;
  if (idx >= total) return;
  int h    = idx & 15;
  int lane = (idx >> 4) & 31;
  int frag = idx >> 9;
  int nt = frag % NT;
  int kc = frag / NT;
  int nl = lane & 15;
  int kl = ((h >> 3) << 4) | ((lane >> 4) << 3) | (h & 7);
  int n = nt * 16 + nl;
  int k = kc * 32 + kl;
  float v = 0.0f;
  if (n < Nreal && k < Kreal) {
    v = src[n * Kstride + k];
    if (mask) v *= mask[k];
  }
  dst[idx] = (_Float16)v;
}

// One workgroup owns 16 batch rows for all T steps. 16 waves: wave w owns
// region-A hidden tile n=w and region-B hidden tile n=w.  x state lives in
// VGPRs; r(t) lives in LDS as ready-to-load WMMA A fragments (double buffer).
__global__ __launch_bounds__(512)
void tworegion_rnn_kernel(const float* __restrict__ inputs,
                          const _Float16* __restrict__ wf,   // staged weights
                          const float* __restrict__ bias_a,
                          const float* __restrict__ bias_b,
                          const float* __restrict__ w_out_b,
                          float* __restrict__ ys,
                          float* __restrict__ xs_a,
                          float* __restrict__ xs_b,
                          float* __restrict__ rs_a,
                          float* __restrict__ rs_b,
                          float* __restrict__ src1,
                          float* __restrict__ src2) {
  // [buf][region][kc*32 + lane] A-fragments of r, 2*2*256*32B = 32 KB LDS
  __shared__ v16h rfragv[2][2][256];

  const int tid  = threadIdx.x;
  const int lane = tid & 31;
  const int wave = tid >> 5;            // 0..15
  const int nl   = lane & 15;
  const int mlo  = (lane >> 4) << 3;    // 0 or 8 (C/D layout row offset)
  const int b0   = blockIdx.x * 16;

  // r(t=-1) = tanh(0) = 0
  {
    _Float16* p = (_Float16*)&rfragv[0][0][0];
    for (int i = tid; i < 2 * 2 * 256 * 16; i += 512) p[i] = (_Float16)0.0f;
  }

  const float bA  = bias_a[wave * 16 + nl];
  const float bB  = bias_b[wave * 16 + nl];
  const float wob = w_out_b[nl & 7];

  // single per-lane weight base; every fragment is reached via an immediate:
  //   matrix AA/AB/BB/BA at OFF_* + kc*8192 halves from this base
  const _Float16* wbase = wf + (size_t)wave * 512 + lane * 16;
  const _Float16* wout  = wf + OFF_OUTW + lane * 16;   // wave 15 only

  v8f xA = {0,0,0,0,0,0,0,0};
  v8f xB = {0,0,0,0,0,0,0,0};

  // constants for scattering r_new (C layout) into A-fragment layout:
  // value (m, k=wave*16+nl): kc=wave>>1, kl=((wave&1)<<4)|nl,
  // h = (kl&7)|((kl>>4)<<3), lane' = m + 16*((kl>>3)&1)
  const int kl_w  = ((wave & 1) << 4) | nl;
  const int kc_w  = wave >> 1;
  const int h_w   = (kl_w & 7) | ((kl_w >> 4) << 3);
  const int lhi_w = ((kl_w >> 3) & 1) << 4;
  const int koff  = (lane >> 4) << 3;   // A-layout K offset for input frag

  __syncthreads();

  int cur = 0;
  for (int t = 0; t < T_STEPS; ++t) {
    // readout ys[t-1] = r_b(t-1) @ w_out^T + b  (wave 15, one step behind)
    if (wave == 15 && t > 0) {
      v8f accY = {0,0,0,0,0,0,0,0};
#pragma unroll 2
      for (int kc = 0; kc < 8; ++kc) {
        v16h a = rfragv[cur][1][kc * 32 + lane];
        v16h b = *(const v16h*)(wout + (size_t)kc * 512);
        accY = wmma16(a, b, accY);
      }
      if (nl < 8) {
        size_t rowBase = (size_t)(t - 1) * BATCH + b0;
#pragma unroll
        for (int v = 0; v < 8; ++v)
          ys[(rowBase + v + mlo) * OUT_O + nl] = accY[v] + wob;
      }
    }

    // build u_t A-fragment (K=32); masks are folded into staged w_in
    const float* urow = inputs + ((size_t)t * BATCH + b0 + nl) * DIN;
    v4f f0 = *(const v4f*)(urow + koff);
    v4f f1 = *(const v4f*)(urow + koff + 4);
    v4f f2 = *(const v4f*)(urow + koff + 16);
    v4f f3 = *(const v4f*)(urow + koff + 20);
    v16h u;
#pragma unroll
    for (int i = 0; i < 4; ++i) {
      u[i]      = (_Float16)f0[i];
      u[4 + i]  = (_Float16)f1[i];
      u[8 + i]  = (_Float16)f2[i];
      u[12 + i] = (_Float16)f3[i];
    }

    v8f z = {0,0,0,0,0,0,0,0};
    // input projections (weights reloaded per step: L0-hot, frees registers)
    v8f accA = wmma16(u, *(const v16h*)(wbase + (OFF_INA - OFF_AA)), z);
    v8f accR = wmma16(u, *(const v16h*)(wbase + (OFF_INB - OFF_AA)), z);
    v8f accS = z;                    // source_b = r_a @ w_ba^T (also an output)

    // partial unroll: bounds in-flight weight fragments -> no scratch spills
#pragma unroll 2
    for (int kc = 0; kc < 8; ++kc) {
      v16h aRA = rfragv[cur][0][kc * 32 + lane];
      v16h aRB = rfragv[cur][1][kc * 32 + lane];
      accA = wmma16(aRA, *(const v16h*)(wbase + OFF_AA + (size_t)kc * 8192), accA); // rec_a
      accA = wmma16(aRB, *(const v16h*)(wbase + OFF_AB + (size_t)kc * 8192), accA); // fb_a
      accR = wmma16(aRB, *(const v16h*)(wbase + OFF_BB + (size_t)kc * 8192), accR); // rec_b
      accS = wmma16(aRA, *(const v16h*)(wbase + OFF_BA + (size_t)kc * 8192), accS); // src_b
    }

    const int nxt = cur ^ 1;
    _Float16* wA = (_Float16*)&rfragv[nxt][0][0] + kc_w * 512 + h_w;
    _Float16* wB = (_Float16*)&rfragv[nxt][1][0] + kc_w * 512 + h_w;
    size_t rowBase = (size_t)t * BATCH + b0;
#pragma unroll
    for (int v = 0; v < 8; ++v) {
      int m = v + mlo;
      size_t g = (rowBase + m) * HID + wave * 16 + nl;
      // region A: x += alpha_a * (-x + rec + fb + in + bias)
      float xa  = xA[v];
      float xan = fmaf(ALPHA_A, (accA[v] + bA) - xa, xa);
      xA[v] = xan;
      float ran = tanhf(xan);
      xs_a[g] = xan;
      rs_a[g] = ran;
      wA[(m + lhi_w) * 16] = (_Float16)ran;
      // region B: x += alpha_b * (-x + rec + src + in + bias)
      float sb  = accS[v];
      float xb  = xB[v];
      float xbn = fmaf(ALPHA_B, (accR[v] + sb + bB) - xb, xb);
      xB[v] = xbn;
      float rbn = tanhf(xbn);
      xs_b[g] = xbn;
      rs_b[g] = rbn;
      src1[g] = sb;
      src2[g] = sb;
      wB[(m + lhi_w) * 16] = (_Float16)rbn;
    }
    __syncthreads();
    cur = nxt;
  }

  // final readout ys[T-1]
  if (wave == 15) {
    v8f accY = {0,0,0,0,0,0,0,0};
#pragma unroll 2
    for (int kc = 0; kc < 8; ++kc) {
      v16h a = rfragv[cur][1][kc * 32 + lane];
      v16h b = *(const v16h*)(wout + (size_t)kc * 512);
      accY = wmma16(a, b, accY);
    }
    if (nl < 8) {
      size_t rowBase = (size_t)(T_STEPS - 1) * BATCH + b0;
#pragma unroll
      for (int v = 0; v < 8; ++v)
        ys[(rowBase + v + mlo) * OUT_O + nl] = accY[v] + wob;
    }
  }
}

extern "C" void kernel_launch(void* const* d_in, const int* in_sizes, int n_in,
                              void* d_out, int out_size, void* d_ws, size_t ws_size,
                              hipStream_t stream) {
  (void)in_sizes; (void)n_in; (void)out_size; (void)ws_size;
  const float* inputs  = (const float*)d_in[0];
  const float* w_aa    = (const float*)d_in[1];
  const float* w_bb    = (const float*)d_in[2];
  const float* w_ab    = (const float*)d_in[3];
  const float* w_ba    = (const float*)d_in[4];
  const float* w_in_a  = (const float*)d_in[5];
  const float* w_in_b  = (const float*)d_in[6];
  const float* bias_a  = (const float*)d_in[7];
  const float* bias_b  = (const float*)d_in[8];
  const float* w_out_w = (const float*)d_in[9];
  const float* w_out_b = (const float*)d_in[10];
  const float* mask_a  = (const float*)d_in[11];
  const float* mask_b  = (const float*)d_in[12];

  // d_ws: staged f16 WMMA B-fragments (~566 KB, always hot in 192MB L2)
  _Float16* ws = (_Float16*)d_ws;

  float* out  = (float*)d_out;
  float* ys   = out;                 // [T,B,8]
  float* xs_a = out + 524288;        // [T,B,256]
  float* xs_b = out + 17301504;
  float* rs_a = out + 34078720;
  float* rs_b = out + 50855936;
  float* src1 = out + 67633152;
  float* src2 = out + 84410368;

  stage_weights_kernel<<<256, 256, 0, stream>>>(w_aa,   ws + OFF_AA,  256, 256, 16, 8, 256, nullptr);
  stage_weights_kernel<<<256, 256, 0, stream>>>(w_ab,   ws + OFF_AB,  256, 256, 16, 8, 256, nullptr);
  stage_weights_kernel<<<256, 256, 0, stream>>>(w_bb,   ws + OFF_BB,  256, 256, 16, 8, 256, nullptr);
  stage_weights_kernel<<<256, 256, 0, stream>>>(w_ba,   ws + OFF_BA,  256, 256, 16, 8, 256, nullptr);
  stage_weights_kernel<<<32,  256, 0, stream>>>(w_in_a, ws + OFF_INA, 256,  32, 16, 1,  32, mask_a);
  stage_weights_kernel<<<32,  256, 0, stream>>>(w_in_b, ws + OFF_INB, 256,  32, 16, 1,  32, mask_b);
  stage_weights_kernel<<<16,  256, 0, stream>>>(w_out_w, ws + OFF_OUTW,  8, 256,  1, 8, 256, nullptr);

  tworegion_rnn_kernel<<<16, 512, 0, stream>>>(
      inputs, ws, bias_a, bias_b, w_out_b,
      ys, xs_a, xs_b, rs_a, rs_b, src1, src2);
}
